// AttModel_24352464569185
// MI455X (gfx1250) — compile-verified, matrix operand order checked by reading, not run on hardware
//
#include <hip/hip_runtime.h>

// ---------------------------------------------------------------------------
// Transformer encoder-decoder forward (AttModel) for gfx1250 (CDNA5).
// All GEMMs run through V_WMMA_F32_16X16X32_F16 (f16 in, f32 accumulate).
// Each wave computes a 16x64 output tile: one A fragment feeds 4 WMMAs per
// K-step (4x less A traffic; all N dims in this model are multiples of 64).
// ---------------------------------------------------------------------------

typedef __attribute__((ext_vector_type(16))) _Float16 v16h;
typedef __attribute__((ext_vector_type(8)))  float    v8f;

#define D_   512
#define H_   8
#define DH_  64
#define NB_  4
#define FF_  2048
#define VOC_ 32000
#define B_   4
#define T_   512
#define M2_  (B_ * T_)     // 2048 rows
#define BH_  (B_ * H_)     // 32 attention batches
#define NEGF (-4294967295.0f)

// ---------------------------------------------------------------------------
// Batched WMMA GEMM:  C[M,N] = act( alpha * A[M,K] @ Bt[N,K]^T + bias )
// A, Bt are f16 row-major; Bt is the transposed B operand ([N,K]).
// One 16(M) x 64(N) tile per wave; K loop in steps of 32. N % 64 == 0.
// ---------------------------------------------------------------------------
union FragH { v16h v; unsigned int u[8]; };

__global__ __launch_bounds__(256) void wmma_gemm_f16(
    const _Float16* __restrict__ A,  long lda, long strideA,
    const _Float16* __restrict__ Bt, long ldb, long strideB,
    float* __restrict__ C,           long ldc, long strideC,
    const float* __restrict__ bias,
    int M, int N, int K, float alpha, int relu)
{
  const int lane  = threadIdx.x & 31;   // wave32
  const int wave  = threadIdx.x >> 5;
  const int wpb   = blockDim.x >> 5;
  const int lo    = lane & 15;
  const int hi    = lane >> 4;          // 0 or 1
  const int kbase = hi * 8;             // lanes 16-31 handle K offsets +8 / +24
  const int mt_n  = M >> 4;
  const int nb_n  = N >> 6;             // 64-wide N blocks
  const int tiles = mt_n * nb_n;

  const long batch = blockIdx.y;
  const _Float16* Ab = A  + batch * strideA;
  const _Float16* Bb = Bt + batch * strideB;
  float*          Cb = C  + batch * strideC;

  for (int t = blockIdx.x * wpb + wave; t < tiles; t += gridDim.x * wpb) {
    const int mt  = t / nb_n;
    const int nb0 = (t - mt * nb_n) * 4;           // first 16-wide N tile
    const _Float16* arow  = Ab + (long)(mt * 16 + lo) * lda;
    const _Float16* brow0 = Bb + (long)(nb0 * 16 + lo) * ldb;
    const _Float16* brow1 = brow0 + 16 * ldb;
    const _Float16* brow2 = brow0 + 32 * ldb;
    const _Float16* brow3 = brow0 + 48 * ldb;

    v8f acc0 = {}, acc1 = {}, acc2 = {}, acc3 = {};
    for (int k0 = 0; k0 < K; k0 += 32) {
      __builtin_prefetch(arow + k0 + 64, 0, 1);    // global_prefetch_b8
      FragH a, b0, b1, b2, b3;
#pragma unroll
      for (int j = 0; j < 8; ++j) {
        // ISA 16-bit A layout: VGPR j<4 -> K = 2j,2j+1 ; j>=4 -> K = 16+2(j-4)
        // lanes 16-31 shifted by +8 (kbase). Each dword = 2 consecutive halves.
        const int koff = k0 + kbase + ((j < 4) ? (2 * j) : (16 + 2 * (j - 4)));
        a.u[j]  = *(const unsigned int*)(arow  + koff);
        b0.u[j] = *(const unsigned int*)(brow0 + koff);
        b1.u[j] = *(const unsigned int*)(brow1 + koff);
        b2.u[j] = *(const unsigned int*)(brow2 + koff);
        b3.u[j] = *(const unsigned int*)(brow3 + koff);
      }
      acc0 = __builtin_amdgcn_wmma_f32_16x16x32_f16(false, a.v, false, b0.v,
                                                    (short)0, acc0, false, false);
      acc1 = __builtin_amdgcn_wmma_f32_16x16x32_f16(false, a.v, false, b1.v,
                                                    (short)0, acc1, false, false);
      acc2 = __builtin_amdgcn_wmma_f32_16x16x32_f16(false, a.v, false, b2.v,
                                                    (short)0, acc2, false, false);
      acc3 = __builtin_amdgcn_wmma_f32_16x16x32_f16(false, a.v, false, b3.v,
                                                    (short)0, acc3, false, false);
    }

#pragma unroll
    for (int r = 0; r < 8; ++r) {
      const int m = mt * 16 + r + 8 * hi;   // C/D layout: VGPR r -> M=r (+8 hi)
      const int n = nb0 * 16 + lo;
      float* crow = Cb + (long)m * ldc + n;
      float v0 = acc0[r] * alpha;
      float v1 = acc1[r] * alpha;
      float v2 = acc2[r] * alpha;
      float v3 = acc3[r] * alpha;
      if (bias) {
        const float* bp = bias + n;
        v0 += bp[0]; v1 += bp[16]; v2 += bp[32]; v3 += bp[48];
      }
      if (relu) {
        v0 = fmaxf(v0, 0.0f); v1 = fmaxf(v1, 0.0f);
        v2 = fmaxf(v2, 0.0f); v3 = fmaxf(v3, 0.0f);
      }
      crow[0] = v0; crow[16] = v1; crow[32] = v2; crow[48] = v3;
    }
  }
}

// ---------------------------------------------------------------------------
// Elementwise / data movement kernels
// ---------------------------------------------------------------------------
__global__ void embed_kernel(const int* __restrict__ tok,
                             const float* __restrict__ emb,
                             const float* __restrict__ pos,
                             float* __restrict__ out,
                             long total, int Tlen, float scale)
{
  long i = (long)blockIdx.x * blockDim.x + threadIdx.x;
  if (i >= total) return;
  long row = i / D_;
  int  d   = (int)(i % D_);
  int  t   = (int)(row % Tlen);
  out[i] = emb[(long)tok[row] * D_ + d] * scale + pos[(long)t * D_ + d];
}

__global__ void f32_to_f16_kernel(const float* __restrict__ in,
                                  _Float16* __restrict__ out, long n)
{
  long i = (long)blockIdx.x * blockDim.x + threadIdx.x;
  if (i < n) out[i] = (_Float16)in[i];
}

// W[K,N] f32 -> Wt[N,K] f16
__global__ void transpose_f32_to_f16(const float* __restrict__ W,
                                     _Float16* __restrict__ Wt,
                                     long total, int K, int N)
{
  long i = (long)blockIdx.x * blockDim.x + threadIdx.x;  // i = n*K + k
  if (i >= total) return;
  int  k = (int)(i % K);
  long n = i / K;
  Wt[i] = (_Float16)W[(long)k * N + n];
}

// X[B*T, D] f32 -> Y[B*H, T, dh] f16
__global__ void split_heads_kernel(const float* __restrict__ X,
                                   _Float16* __restrict__ Y, long total)
{
  long i = (long)blockIdx.x * blockDim.x + threadIdx.x;
  if (i >= total) return;
  int  d  = (int)(i % DH_);
  long r  = i / DH_;
  int  t  = (int)(r % T_);
  long bh = r / T_;
  long b  = bh / H_;
  long h  = bh % H_;
  Y[i] = (_Float16)X[(b * T_ + t) * (long)D_ + h * DH_ + d];
}

// X[B*T, D] f32 -> Y[B*H, dh, T] f16  (transposed per head, for the AV B-operand)
__global__ void split_heads_T_kernel(const float* __restrict__ X,
                                     _Float16* __restrict__ Y, long total)
{
  long i = (long)blockIdx.x * blockDim.x + threadIdx.x;
  if (i >= total) return;
  int  t  = (int)(i % T_);
  long r  = i / T_;
  int  d  = (int)(r % DH_);
  long bh = r / DH_;
  long b  = bh / H_;
  long h  = bh % H_;
  Y[i] = (_Float16)X[(b * T_ + t) * (long)D_ + h * DH_ + d];
}

// Y[B*H, T, dh] f32 -> X[B*T, D] f32
__global__ void merge_heads_kernel(const float* __restrict__ Y,
                                   float* __restrict__ X, long total)
{
  long i = (long)blockIdx.x * blockDim.x + threadIdx.x;
  if (i >= total) return;
  int  c = (int)(i % D_);
  long r = i / D_;          // b*T + t
  int  t = (int)(r % T_);
  long b = r / T_;
  int  h = c / DH_;
  int  d = c % DH_;
  X[i] = Y[((b * H_ + h) * (long)T_ + t) * DH_ + d];
}

// row mask = sign(abs(sum over channels))  -> 0.0 or 1.0
__global__ __launch_bounds__(256) void row_mask_kernel(
    const float* __restrict__ X, float* __restrict__ m)
{
  __shared__ float red[256];
  long row = blockIdx.x;
  const float* x = X + row * D_;
  int tid = threadIdx.x;
  red[tid] = x[tid] + x[tid + 256];
  __syncthreads();
  for (int s = 128; s > 0; s >>= 1) {
    if (tid < s) red[tid] += red[tid + s];
    __syncthreads();
  }
  if (tid == 0) m[row] = (red[0] != 0.0f) ? 1.0f : 0.0f;
}

// masked, numerically-stable softmax over one row of scores; writes f16 probs
__global__ __launch_bounds__(512) void attn_softmax_kernel(
    const float* __restrict__ S, _Float16* __restrict__ P,
    const float* __restrict__ km, const float* __restrict__ qm, int causal)
{
  __shared__ float red[512];
  const long row = blockIdx.x;            // over B*H*Tq
  const int  q   = (int)(row % T_);
  const long bh  = row / T_;
  const long b   = bh / H_;
  const int  k   = threadIdx.x;

  float v = S[row * (long)T_ + k];
  if (km[b * T_ + k] == 0.0f) v = NEGF;
  if (causal && k > q)        v = NEGF;

  red[k] = v;
  __syncthreads();
  for (int s = 256; s > 0; s >>= 1) {
    if (k < s) red[k] = fmaxf(red[k], red[k + s]);
    __syncthreads();
  }
  const float mx = red[0];
  __syncthreads();

  const float e = __expf(v - mx);
  red[k] = e;
  __syncthreads();
  for (int s = 256; s > 0; s >>= 1) {
    if (k < s) red[k] += red[k + s];
    __syncthreads();
  }
  const float sum = red[0];
  const float p = (e / sum) * qm[b * T_ + q];   // query mask AFTER softmax
  P[row * (long)T_ + k] = (_Float16)p;
}

// out = g * (X + R - mean) / (std_bessel + eps) + beta   (row = 512 channels)
__global__ __launch_bounds__(256) void residual_layernorm_kernel(
    const float* __restrict__ X, const float* __restrict__ R,
    const float* __restrict__ g, const float* __restrict__ beta,
    float* __restrict__ O)
{
  __shared__ float red[256];
  const long row = blockIdx.x;
  const float* x = X + row * D_;
  const float* r = R + row * D_;
  float*       o = O + row * D_;
  const int tid = threadIdx.x;

  const float v0 = x[tid] + r[tid];
  const float v1 = x[tid + 256] + r[tid + 256];

  red[tid] = v0 + v1;
  __syncthreads();
  for (int s = 128; s > 0; s >>= 1) {
    if (tid < s) red[tid] += red[tid + s];
    __syncthreads();
  }
  const float mean = red[0] / (float)D_;
  __syncthreads();

  const float d0 = v0 - mean, d1 = v1 - mean;
  red[tid] = d0 * d0 + d1 * d1;
  __syncthreads();
  for (int s = 128; s > 0; s >>= 1) {
    if (tid < s) red[tid] += red[tid + s];
    __syncthreads();
  }
  const float stdv = sqrtf(red[0] / (float)(D_ - 1));   // ddof = 1 (Bessel)
  const float inv  = 1.0f / (stdv + 1e-8f);

  o[tid]       = g[tid]       * d0 * inv + beta[tid];
  o[tid + 256] = g[tid + 256] * d1 * inv + beta[tid + 256];
}

// ---------------------------------------------------------------------------
// Host orchestration
// ---------------------------------------------------------------------------
struct LinP { const float* W; const float* b; };
struct LNP  { const float* g; const float* b; };
struct AttP { LinP q, k, v; LNP ln; };
struct FFP  { LinP l1, l2; LNP ln; };

extern "C" void kernel_launch(void* const* d_in, const int* in_sizes, int n_in,
                              void* d_out, int out_size, void* d_ws, size_t ws_size,
                              hipStream_t stream)
{
  (void)in_sizes; (void)n_in; (void)out_size; (void)ws_size;

  // ---- inputs in setup_inputs() dict (insertion) order --------------------
  int p = 0;
  auto F = [&](void) { return (const float*)d_in[p++]; };
  const int* xtok = (const int*)d_in[p++];
  const int* ytok = (const int*)d_in[p++];
  const float* enc_emb = F();
  const float* dec_emb = F();
  const float* enc_pos = F();
  const float* dec_pos = F();
  auto lin = [&](void) { LinP l; l.W = F(); l.b = F(); return l; };
  auto lnp = [&](void) { LNP l; l.g = F(); l.b = F(); return l; };
  auto att = [&](void) { AttP a; a.q = lin(); a.k = lin(); a.v = lin(); a.ln = lnp(); return a; };
  auto ffp = [&](void) { FFP f; f.l1 = lin(); f.l2 = lin(); f.ln = lnp(); return f; };

  AttP enc_self[NB_]; FFP enc_ff[NB_];
  for (int i = 0; i < NB_; ++i) { enc_self[i] = att(); enc_ff[i] = ffp(); }
  AttP dec_self[NB_], dec_cross[NB_]; FFP dec_ff[NB_];
  for (int i = 0; i < NB_; ++i) { dec_self[i] = att(); dec_cross[i] = att(); dec_ff[i] = ffp(); }
  LinP outp = lin();

  // ---- workspace bump allocator ------------------------------------------
  char* wsb = (char*)d_ws;
  size_t off = 0;
  auto alloc = [&](size_t bytes) {
    off = (off + 255) & ~(size_t)255;
    void* ptr = wsb + off;
    off += bytes;
    return ptr;
  };
  float*     enc = (float*)alloc((size_t)M2_ * D_ * 4);
  float*     dec = (float*)alloc((size_t)M2_ * D_ * 4);
  float*     qf  = (float*)alloc((size_t)M2_ * D_ * 4);
  float*     kf  = (float*)alloc((size_t)M2_ * D_ * 4);
  float*     vf  = (float*)alloc((size_t)M2_ * D_ * 4);
  float*     t0  = (float*)alloc((size_t)M2_ * D_ * 4);
  float*     hf  = (float*)alloc((size_t)M2_ * FF_ * 4);
  float*     S   = (float*)alloc((size_t)BH_ * T_ * T_ * 4);
  float*     oc  = (float*)alloc((size_t)BH_ * T_ * DH_ * 4);
  float*     qmb = (float*)alloc((size_t)M2_ * 4);
  float*     kmb = (float*)alloc((size_t)M2_ * 4);
  _Float16*  ah  = (_Float16*)alloc((size_t)M2_ * D_ * 2);
  _Float16*  ah2 = (_Float16*)alloc((size_t)M2_ * D_ * 2);
  _Float16*  qh  = (_Float16*)alloc((size_t)BH_ * T_ * DH_ * 2);
  _Float16*  kh  = (_Float16*)alloc((size_t)BH_ * T_ * DH_ * 2);
  _Float16*  vt  = (_Float16*)alloc((size_t)BH_ * DH_ * T_ * 2);
  _Float16*  ph  = (_Float16*)alloc((size_t)BH_ * T_ * T_ * 2);
  _Float16*  hh  = (_Float16*)alloc((size_t)M2_ * FF_ * 2);
  _Float16*  wt  = (_Float16*)alloc((size_t)D_ * VOC_ * 2);  // largest weight

  // ---- launch helpers -----------------------------------------------------
  auto gemm = [&](const _Float16* A, long lda, long sA,
                  const _Float16* Bt, long ldb, long sB,
                  float* C, long ldc, long sC, const float* bias,
                  int M, int N, int K, float alpha, int relu, int batches) {
    int tiles  = (M / 16) * (N / 64);       // 16x64 tile per wave
    int blocks = (tiles + 7) / 8;           // 8 waves per 256-thread block
    if (blocks < 1) blocks = 1;
    if (blocks > 16384) blocks = 16384;     // grid-stride covers the rest
    wmma_gemm_f16<<<dim3(blocks, batches), dim3(256), 0, stream>>>(
        A, lda, sA, Bt, ldb, sB, C, ldc, sC, bias, M, N, K, alpha, relu);
  };
  auto trW = [&](const float* W, int K, int N) {   // -> wt = W^T as f16 [N,K]
    long n = (long)K * N;
    transpose_f32_to_f16<<<(int)((n + 255) / 256), 256, 0, stream>>>(W, wt, n, K, N);
  };
  auto toH = [&](const float* X, _Float16* Y, long n) {
    f32_to_f16_kernel<<<(int)((n + 255) / 256), 256, 0, stream>>>(X, Y, n);
  };

  const long nSplit = (long)BH_ * T_ * DH_;
  const long nMerge = (long)M2_ * D_;

  auto mha = [&](float* q_in, float* kv_in, const AttP& A, int causal) {
    // masks from pre-projection activations
    row_mask_kernel<<<M2_, 256, 0, stream>>>(q_in, qmb);
    row_mask_kernel<<<M2_, 256, 0, stream>>>(kv_in, kmb);
    // f16 GEMM inputs
    toH(q_in, ah,  (long)M2_ * D_);
    toH(kv_in, ah2, (long)M2_ * D_);
    // Q/K/V projections (Linear + ReLU)
    trW(A.q.W, D_, D_);
    gemm(ah, D_, 0, wt, D_, 0, qf, D_, 0, A.q.b, M2_, D_, D_, 1.0f, 1, 1);
    trW(A.k.W, D_, D_);
    gemm(ah2, D_, 0, wt, D_, 0, kf, D_, 0, A.k.b, M2_, D_, D_, 1.0f, 1, 1);
    trW(A.v.W, D_, D_);
    gemm(ah2, D_, 0, wt, D_, 0, vf, D_, 0, A.v.b, M2_, D_, D_, 1.0f, 1, 1);
    // head split (f16)
    split_heads_kernel<<<(int)((nSplit + 255) / 256), 256, 0, stream>>>(qf, qh, nSplit);
    split_heads_kernel<<<(int)((nSplit + 255) / 256), 256, 0, stream>>>(kf, kh, nSplit);
    split_heads_T_kernel<<<(int)((nSplit + 255) / 256), 256, 0, stream>>>(vf, vt, nSplit);
    // S = (Q K^T) / sqrt(dh) ; Kh row-major [T,dh] IS the transposed B operand
    gemm(qh, DH_, (long)T_ * DH_, kh, DH_, (long)T_ * DH_,
         S, T_, (long)T_ * T_, nullptr, T_, T_, DH_, 0.125f, 0, BH_);
    // masked softmax -> f16 probabilities
    attn_softmax_kernel<<<BH_ * T_, T_, 0, stream>>>(S, ph, kmb, qmb, causal);
    // O = P @ V  (Bt = V^T per head)
    gemm(ph, T_, (long)T_ * T_, vt, T_, (long)DH_ * T_,
         oc, DH_, (long)T_ * DH_, nullptr, T_, DH_, T_, 1.0f, 0, BH_);
    // merge heads, residual + layernorm (in place on q_in)
    merge_heads_kernel<<<(int)((nMerge + 255) / 256), 256, 0, stream>>>(oc, t0, nMerge);
    residual_layernorm_kernel<<<M2_, 256, 0, stream>>>(t0, q_in, A.ln.g, A.ln.b, q_in);
  };

  auto ffn = [&](float* xbuf, const FFP& Fp) {
    toH(xbuf, ah, (long)M2_ * D_);
    trW(Fp.l1.W, D_, FF_);
    gemm(ah, D_, 0, wt, D_, 0, hf, FF_, 0, Fp.l1.b, M2_, FF_, D_, 1.0f, 1, 1);
    toH(hf, hh, (long)M2_ * FF_);
    trW(Fp.l2.W, FF_, D_);
    gemm(hh, FF_, 0, wt, FF_, 0, t0, D_, 0, Fp.l2.b, M2_, D_, FF_, 1.0f, 0, 1);
    residual_layernorm_kernel<<<M2_, 256, 0, stream>>>(t0, xbuf, Fp.ln.g, Fp.ln.b, xbuf);
  };

  // ---- encoder ------------------------------------------------------------
  const float scale = sqrtf((float)D_);
  const long nEmb = (long)M2_ * D_;
  embed_kernel<<<(int)((nEmb + 255) / 256), 256, 0, stream>>>(
      xtok, enc_emb, enc_pos, enc, nEmb, T_, scale);
  for (int i = 0; i < NB_; ++i) {
    mha(enc, enc, enc_self[i], /*causal=*/0);
    ffn(enc, enc_ff[i]);
  }

  // ---- decoder ------------------------------------------------------------
  embed_kernel<<<(int)((nEmb + 255) / 256), 256, 0, stream>>>(
      ytok, dec_emb, dec_pos, dec, nEmb, T_, scale);
  for (int i = 0; i < NB_; ++i) {
    mha(dec, dec, dec_self[i],  /*causal=*/1);
    mha(dec, enc, dec_cross[i], /*causal=*/0);
    ffn(dec, dec_ff[i]);
  }

  // ---- output projection to logits ---------------------------------------
  toH(dec, ah, (long)M2_ * D_);
  trW(outp.W, D_, VOC_);
  gemm(ah, D_, 0, wt, D_, 0, (float*)d_out, VOC_, 0, outp.b,
       M2_, VOC_, D_, 1.0f, 0, 1);
}
